// GCN_22385369547105
// MI455X (gfx1250) — compile-verified, matrix operand order checked by reading, not run on hardware
//
#include <hip/hip_runtime.h>

// GCN forward on gfx1250 (MI455X):
//   deg -> c_src/c_dst = rsqrt(max(deg,1))
//   T0   = (x * c_src) @ W0                                 (bf16 WMMA, fused scale)
//   AGG0 = scatter_add(T0[src] -> dst)                      (f32 atomics, L2-resident)
//   T1   = (relu(AGG0*c_dst + b0) * c_src) @ W1             (bf16 WMMA, fused epilogue/prologue)
//   AGG1 = scatter_add(T1[src] -> dst)
//   out  = relu(AGG1*c_dst + b1) @ Wc + bc                  (bf16 WMMA, 47-col masked)

#define KDIM 128
#define HID  128
#define NCLS 47

typedef __attribute__((ext_vector_type(16))) __bf16 v16bf;
typedef __attribute__((ext_vector_type(8)))  float  v8f;

__global__ void zero_f32(float* __restrict__ p, long long n) {
  long long i      = (long long)blockIdx.x * blockDim.x + threadIdx.x;
  long long stride = (long long)gridDim.x * blockDim.x;
  long long n4     = n >> 2;
  float4*   p4     = (float4*)p;
  const float4 z   = {0.0f, 0.0f, 0.0f, 0.0f};
  for (long long j = i; j < n4; j += stride) p4[j] = z;
  for (long long j = (n4 << 2) + i; j < n; j += stride) p[j] = 0.0f;
}

__global__ void degree_kernel(const int* __restrict__ edges, float* __restrict__ outdeg,
                              float* __restrict__ indeg, int E) {
  int e = blockIdx.x * blockDim.x + threadIdx.x;
  if (e >= E) return;
  atomicAdd(&outdeg[edges[e]], 1.0f);       // src out-degree
  atomicAdd(&indeg[edges[E + e]], 1.0f);    // dst in-degree
}

__global__ void invsqrt_kernel(float* __restrict__ c, int n) {
  int i = blockIdx.x * blockDim.x + threadIdx.x;
  if (i >= n) return;
  float d = c[i];
  d = d < 1.0f ? 1.0f : d;
  c[i] = rsqrtf(d);
}

// One wave per edge; lane handles 4 consecutive feature floats.
__global__ void scatter_kernel(const int* __restrict__ edges, const float* __restrict__ T,
                               float* __restrict__ AGG, int E) {
  int tid  = blockIdx.x * blockDim.x + threadIdx.x;
  int e    = tid >> 5;
  int lane = tid & 31;
  if (e >= E) return;
  int s = edges[e];
  int d = edges[E + e];
  float4 v = *((const float4*)(T + (size_t)s * HID) + lane);
  float* o = AGG + (size_t)d * HID + lane * 4;
  atomicAdd(o + 0, v.x);
  atomicAdd(o + 1, v.y);
  atomicAdd(o + 2, v.z);
  atomicAdd(o + 3, v.w);
}

// PROLOG 0: a = A[r][k] * csrc[r]
// PROLOG 1: a = relu(A[r][k]*cdst[r] + bias_in[k]) * csrc[r]
// PROLOG 2: a = relu(A[r][k]*cdst[r] + bias_in[k])           (classifier input)
// One wave -> one 16x16 C tile; K = 128 -> 4x v_wmma_f32_16x16x32_bf16.
// Out-of-range rows/cols are CLAMPED on load (they only feed C rows/cols that
// are masked at the store), so fragment loads are branch-free and vectorized.
template <int PROLOG, bool BIAS_OUT, int NB>
__global__ void gemm_bf16_wmma(const float* __restrict__ A, const float* __restrict__ B,
                               float* __restrict__ Out,
                               const float* __restrict__ csrc, const float* __restrict__ cdst,
                               const float* __restrict__ bias_in, const float* __restrict__ bias_out,
                               int N) {
  const int lane    = threadIdx.x & 31;
  const int wave    = threadIdx.x >> 5;
  const int rowBase = blockIdx.x << 4;
  const int colBase = wave << 4;
  const int half    = lane & 15;
  const int kgrp    = (lane >> 4) << 3;   // 0 or 8 (K sub-group per half-wave)

  int aRow = rowBase + half;
  aRow = aRow < N ? aRow : N - 1;         // clamp (store is masked)
  int bCol = colBase + half;
  bCol = bCol < NB ? bCol : NB - 1;       // clamp (store is masked)

  const float csrc_r = (PROLOG <= 1) ? csrc[aRow] : 0.0f;
  const float cdst_r = (PROLOG >= 1) ? cdst[aRow] : 0.0f;

  const float* aPtr = A + (size_t)aRow * KDIM;

  v8f acc = {};

#pragma unroll
  for (int kk = 0; kk < KDIM; kk += 32) {
    const int k0 = kk + kgrp;        // K block [k0, k0+7]  (32B-aligned)
    const int k1 = kk + 16 + kgrp;   // K block [k1, k1+7]  (32B-aligned)

    // A fragment: two 8-float vector loads (-> 4x global_load_b128)
    const v8f alo = *(const v8f*)(aPtr + k0);
    const v8f ahi = *(const v8f*)(aPtr + k1);

    // bias fragment (layer-1 / classifier prologue only)
    v8f blo = {}, bhi = {};
    if (PROLOG >= 1) {
      blo = *(const v8f*)(bias_in + k0);
      bhi = *(const v8f*)(bias_in + k1);
    }

    // B fragment: 16 column-strided scalar loads (W is tiny -> L1/L2 resident)
    float bv[16];
    const float* bp0 = B + (size_t)k0 * NB + bCol;
    const float* bp1 = B + (size_t)k1 * NB + bCol;
#pragma unroll
    for (int i = 0; i < 8; ++i) {
      bv[i]     = bp0[(size_t)i * NB];
      bv[8 + i] = bp1[(size_t)i * NB];
    }

    v16bf a, b;
#pragma unroll
    for (int i = 0; i < 8; ++i) {
      float va = alo[i];
      float vb = ahi[i];
      if (PROLOG >= 1) {
        va = fmaf(va, cdst_r, blo[i]); va = va > 0.0f ? va : 0.0f;  // relu
        vb = fmaf(vb, cdst_r, bhi[i]); vb = vb > 0.0f ? vb : 0.0f;
      }
      if (PROLOG <= 1) { va *= csrc_r; vb *= csrc_r; }
      a[i]     = (__bf16)va;
      a[8 + i] = (__bf16)vb;
      b[i]     = (__bf16)bv[i];
      b[8 + i] = (__bf16)bv[8 + i];
    }
    acc = __builtin_amdgcn_wmma_f32_16x16x32_bf16(false, a, false, b, (short)0, acc,
                                                  false, false);
  }

  // C/D layout: lane&15 = column, VGPR r -> row r + 8*(lane>>4)
  const int oCol = colBase + half;
  if (oCol < NB) {
    const float bo  = BIAS_OUT ? bias_out[oCol] : 0.0f;
    const int  rOff = rowBase + ((lane >> 4) << 3);
#pragma unroll
    for (int r = 0; r < 8; ++r) {
      const int oRow = rOff + r;
      if (oRow < N) Out[(size_t)oRow * NB + oCol] = acc[r] + bo;
    }
  }
}

extern "C" void kernel_launch(void* const* d_in, const int* in_sizes, int n_in,
                              void* d_out, int out_size, void* d_ws, size_t ws_size,
                              hipStream_t stream) {
  const float* x     = (const float*)d_in[0];
  const int*   edges = (const int*)d_in[1];
  const float* W0    = (const float*)d_in[2];
  const float* b0    = (const float*)d_in[3];
  const float* W1    = (const float*)d_in[4];
  const float* b1    = (const float*)d_in[5];
  const float* Wc    = (const float*)d_in[6];
  const float* bc    = (const float*)d_in[7];
  float*       out   = (float*)d_out;

  const int N = in_sizes[0] / KDIM;
  const int E = in_sizes[1] / 2;

  // Workspace layout (floats): c_src[N] | c_dst[N] | bufA[N*128] | bufB[N*128]  (~103 MB)
  float* ws    = (float*)d_ws;
  float* c_src = ws;
  float* c_dst = ws + (size_t)N;
  float* bufA  = ws + 2 * (size_t)N;
  float* bufB  = bufA + (size_t)N * HID;

  const int rowTiles   = (N + 15) / 16;
  const int edgeBlocks = (E + 255) / 256;
  const int scatBlocks = (int)(((long long)E * 32 + 255) / 256);

  // --- degrees -> symmetric normalization coefficients ---
  zero_f32<<<1024, 256, 0, stream>>>(c_src, 2LL * N);
  degree_kernel<<<edgeBlocks, 256, 0, stream>>>(edges, c_src, c_dst, E);
  invsqrt_kernel<<<(2 * N + 255) / 256, 256, 0, stream>>>(c_src, 2 * N);

  // --- layer 0: T0 = (x * c_src) @ W0 ; AGG0 = scatter(T0) ---
  gemm_bf16_wmma<0, false, HID><<<rowTiles, 256, 0, stream>>>(
      x, W0, bufA, c_src, nullptr, nullptr, nullptr, N);
  zero_f32<<<4096, 256, 0, stream>>>(bufB, (long long)N * HID);
  scatter_kernel<<<scatBlocks, 256, 0, stream>>>(edges, bufA, bufB, E);

  // --- layer 1: T1 = (relu(AGG0*c_dst + b0) * c_src) @ W1 ; AGG1 = scatter(T1) ---
  gemm_bf16_wmma<1, false, HID><<<rowTiles, 256, 0, stream>>>(
      bufB, W1, bufA, c_src, c_dst, b0, nullptr, N);
  zero_f32<<<4096, 256, 0, stream>>>(bufB, (long long)N * HID);
  scatter_kernel<<<scatBlocks, 256, 0, stream>>>(edges, bufA, bufB, E);

  // --- classifier: out = relu(AGG1*c_dst + b1) @ Wc + bc  (47 cols -> 3 waves/block) ---
  gemm_bf16_wmma<2, true, NCLS><<<rowTiles, 96, 0, stream>>>(
      bufB, Wc, out, nullptr, c_dst, b1, bc, N);
}
